// EvenOddFunctionHAM_40888088658618
// MI455X (gfx1250) — matrix operand order repaired and checked by reading, not compiled, not run on hardware
//
#include <hip/hip_runtime.h>

// ---------------------------------------------------------------------------
// EvenOdd HAM forward on MI455X (gfx1250, wave32, WMMA).
//   T   = rho(s) @ W + b_odd ;  T[:, :2048] += Ux ;  out = rho(T) @ W.T + b_even
// bf16 WMMA (v_wmma_f32_16x16x32_bf16) with fp32 accum, block-sparse K ranges
// from the staircase mask, double-buffered async (ASYNCcnt) LDS staging.
// Wave tile 64x64 (16 WMMA / 16 ds_load_b128 per K-step) to match LDS bank
// bandwidth (~256 B/clk/WGP) against the XDL pipes.
// Workspace layout (needs 128 MB):
//   [0,32MB)   Wb  : bf16 W (masked), row-major   -> B^T operand, GEMM2
//   [32,64MB)  Wbt : bf16 W^T (masked), row-major -> B^T operand, GEMM1
//   [64,96MB)  A1  : bf16 rho(s)                  -> A operand, GEMM1
//   [96,128MB) A2  : bf16 rho(T)                  -> A operand, GEMM2
// ---------------------------------------------------------------------------

typedef __attribute__((ext_vector_type(16))) __bf16         v16bf;
typedef __attribute__((ext_vector_type(8)))  float          v8f;
typedef __attribute__((ext_vector_type(4)))  unsigned int   u32x4;
typedef __attribute__((ext_vector_type(4)))  int            i32x4;
typedef __attribute__((ext_vector_type(4)))  unsigned short u16x4;

struct Pack32 { u32x4 lo; u32x4 hi; };   // 32 bytes -> bit_cast to v16bf

#define NDIM 4096
#define D1   2048
#define LDSS 40        // LDS row stride in ushorts (32 data + 8 pad)
#define BM   128       // block tile M
#define BN   256       // block tile N

// ---- CDNA5 async global->LDS DMA (ASYNCcnt), guarded for compile safety ----
#if defined(__has_builtin)
#  if __has_builtin(__builtin_amdgcn_global_load_async_to_lds_b128)
#    define HAVE_ASYNC_LDS 1
#  endif
#endif
#ifndef HAVE_ASYNC_LDS
#  define HAVE_ASYNC_LDS 0
#endif

#if HAVE_ASYNC_LDS
// Builtin signature (probe-confirmed): (int4 AS1*, int4 AS3*, imm off, imm cpol)
typedef __attribute__((address_space(1))) i32x4 as1_i32x4;
typedef __attribute__((address_space(3))) i32x4 as3_i32x4;
#endif

__device__ __forceinline__ void copy16_to_lds(const unsigned short* g,
                                              unsigned short* l) {
#if HAVE_ASYNC_LDS
    __builtin_amdgcn_global_load_async_to_lds_b128(
        (as1_i32x4*)g, (as3_i32x4*)l, 0, 0);
#else
    *reinterpret_cast<u32x4*>(l) = *reinterpret_cast<const u32x4*>(g);
#endif
}

__device__ __forceinline__ void wait_async_done() {
#if HAVE_ASYNC_LDS
#  if __has_builtin(__builtin_amdgcn_s_wait_asynccnt)
    __builtin_amdgcn_s_wait_asynccnt(0);
#  else
    asm volatile("s_wait_asynccnt 0x0" ::: "memory");
#  endif
#endif
}

// ---- fast rho: sigmoid(4x-2) = rcp(1 + exp2((2-4x)*log2e)) -----------------
__device__ __forceinline__ float rho_f(float x) {
    const float e = __builtin_amdgcn_exp2f((2.0f - 4.0f * x) * 1.44269504088896f);
    return __builtin_amdgcn_rcpf(1.0f + e);
}

__device__ __forceinline__ unsigned short f2bf(float f) {
    unsigned int u = __float_as_uint(f);
    u += 0x7FFFu + ((u >> 16) & 1u);     // round-to-nearest-even
    return (unsigned short)(u >> 16);
}

// ---- prep: Wb = bf16(W*mask) row-major, Wbt = transpose; LDS-tiled ---------
__global__ __launch_bounds__(256)
void prep_w_kernel(const float* __restrict__ Wt,
                   const float* __restrict__ Wm,
                   unsigned short* __restrict__ Wb,
                   unsigned short* __restrict__ Wbt) {
    __shared__ unsigned short tile[32][33];   // [orig col][orig row], padded
    const int tx = threadIdx.x;
    const int r0 = blockIdx.y * 32;
    const int c0 = blockIdx.x * 32;

    // Phase 1: coalesced read of 32x32 floats; coalesced Wb write; LDS transpose.
    {
        const int idx = tx * 4;          // 0..1023
        const int r = idx >> 5;          // 0..31
        const int c = idx & 31;
        const size_t g = (size_t)(r0 + r) * NDIM + (c0 + c);
        const float4 w = *reinterpret_cast<const float4*>(&Wt[g]);
        const float4 m = *reinterpret_cast<const float4*>(&Wm[g]);
        u16x4 b;
        b.x = f2bf(w.x * m.x); b.y = f2bf(w.y * m.y);
        b.z = f2bf(w.z * m.z); b.w = f2bf(w.w * m.w);
        *reinterpret_cast<u16x4*>(&Wb[g]) = b;
        tile[c + 0][r] = b.x; tile[c + 1][r] = b.y;
        tile[c + 2][r] = b.z; tile[c + 3][r] = b.w;
    }
    __syncthreads();

    // Phase 2: coalesced Wbt write (4 ushorts along original-row direction).
    {
        const int idx = tx * 4;
        const int c = idx >> 5;          // original col
        const int r = idx & 31;          // original row
        u16x4 b;
        b.x = tile[c][r + 0]; b.y = tile[c][r + 1];
        b.z = tile[c][r + 2]; b.w = tile[c][r + 3];
        *reinterpret_cast<u16x4*>(&Wbt[(size_t)(c0 + c) * NDIM + (r0 + r)]) = b;
    }
}

// ---- prep: A1 = bf16(rho(s)) ------------------------------------------------
__global__ __launch_bounds__(256)
void prep_a_kernel(const float* __restrict__ s,
                   unsigned short* __restrict__ A1) {
    const int i = (blockIdx.x * blockDim.x + threadIdx.x) * 4;
    const float4 v = *reinterpret_cast<const float4*>(s + i);
    u16x4 b;
    b.x = f2bf(rho_f(v.x)); b.y = f2bf(rho_f(v.y));
    b.z = f2bf(rho_f(v.z)); b.w = f2bf(rho_f(v.w));
    *reinterpret_cast<u16x4*>(&A1[i]) = b;
}

// ---- WMMA GEMM: C[m,n] = sum_k A[m,k] * Bt[n,k]  (+ fused epilogue) --------
// Block tile 128(M) x 256(N); 8 waves as 2x4; wave tile 64x64 (acc[4][4]).
// MODE 1: +b_odd, +Ux on n<D1 (block-uniform), rho, write bf16 -> A2
// MODE 2: +b_even, write fp32                                  -> out
template <int MODE>
__global__ __launch_bounds__(256)
void gemm_kernel(const unsigned short* __restrict__ A,    // bf16 [4096 x 4096]
                 const unsigned short* __restrict__ Bt,   // bf16 [4096 x 4096] (B^T)
                 const float* __restrict__ bias,          // [4096]
                 const float* __restrict__ Ux,            // [4096 x 2048] (MODE 1)
                 unsigned short* __restrict__ outBf,      // MODE 1
                 float* __restrict__ outF,                // MODE 2
                 int kbeg_lo, int kend_lo, int kbeg_hi, int kend_hi) {
    __shared__ unsigned short lsA[2][BM * LDSS];   // 2 x 10 KB
    __shared__ unsigned short lsB[2][BN * LDSS];   // 2 x 20 KB

    const int tid  = threadIdx.x;
    const int wave = tid >> 5;
    const int lane = tid & 31;
    const int half = lane >> 4;   // K half-select (WMMA lane layout)
    const int rr   = lane & 15;   // row (A) / col (B) within 16x16 tile

    const int wm = wave >> 2;     // 0..1 -> 64 output rows per wave
    const int wn = wave & 3;      // 0..3 -> 64 output cols per wave

    const int row0 = blockIdx.x * BM;
    const int col0 = blockIdx.y * BN;

    // Staircase sparsity: per-column-block K range (block-uniform; BN | D1).
    const int kbeg = (col0 >= D1) ? kbeg_hi : kbeg_lo;
    const int kend = (col0 >= D1) ? kend_hi : kend_lo;

    // Cooperative bf16 tile stage: A 128x32 (2 chunks/thr), B 256x32 (4 chunks/thr).
    auto load_tile = [&](int ib, int k0) {
#pragma unroll
        for (int i = 0; i < 2; ++i) {
            const int t = tid + i * 256;
            const int r = t >> 2;
            const int c = (t & 3) << 3;
            copy16_to_lds(&A[(size_t)(row0 + r) * NDIM + k0 + c],
                          &lsA[ib][r * LDSS + c]);
        }
#pragma unroll
        for (int i = 0; i < 4; ++i) {
            const int t = tid + i * 256;
            const int r = t >> 2;
            const int c = (t & 3) << 3;
            copy16_to_lds(&Bt[(size_t)(col0 + r) * NDIM + k0 + c],
                          &lsB[ib][r * LDSS + c]);
        }
    };

    v8f acc[4][4];
#pragma unroll
    for (int mt = 0; mt < 4; ++mt)
#pragma unroll
        for (int nt = 0; nt < 4; ++nt)
            acc[mt][nt] = (v8f){0.f, 0.f, 0.f, 0.f, 0.f, 0.f, 0.f, 0.f};

    load_tile(0, kbeg);                       // prologue prefetch
    int ib = 0;
    for (int k0 = kbeg; k0 < kend; k0 += 32, ib ^= 1) {
        wait_async_done();                    // my DMA into buf ib landed
        __syncthreads();                      // everyone's landed / prev reads done

        if (k0 + 32 < kend) load_tile(ib ^ 1, k0 + 32);   // overlap next tile

        // Per-lane WMMA fragments from buffer ib.
        v16bf aF[4];
#pragma unroll
        for (int mt = 0; mt < 4; ++mt) {
            const int off = (wm * 64 + mt * 16 + rr) * LDSS + half * 8;
            Pack32 p;
            p.lo = *reinterpret_cast<const u32x4*>(&lsA[ib][off]);       // K h*8..
            p.hi = *reinterpret_cast<const u32x4*>(&lsA[ib][off + 16]);  // K 16+h*8..
            aF[mt] = __builtin_bit_cast(v16bf, p);
        }
        v16bf bF[4];
#pragma unroll
        for (int nt = 0; nt < 4; ++nt) {
            const int off = (wn * 64 + nt * 16 + rr) * LDSS + half * 8;
            Pack32 p;
            p.lo = *reinterpret_cast<const u32x4*>(&lsB[ib][off]);       // K h*8..
            p.hi = *reinterpret_cast<const u32x4*>(&lsB[ib][off + 16]);  // K 16+h*8..
            bF[nt] = __builtin_bit_cast(v16bf, p);
        }

#pragma unroll
        for (int mt = 0; mt < 4; ++mt)
#pragma unroll
            for (int nt = 0; nt < 4; ++nt)
                acc[mt][nt] = __builtin_amdgcn_wmma_f32_16x16x32_bf16(
                    false, aF[mt], false, bF[nt], (short)0, acc[mt][nt],
                    false, false);
    }

    // Epilogue. C layout: lane -> n = rr, VGPR v -> m = v + half*8.
#pragma unroll
    for (int mt = 0; mt < 4; ++mt) {
#pragma unroll
        for (int nt = 0; nt < 4; ++nt) {
            const int n     = col0 + wn * 64 + nt * 16 + rr;
            const int mbase = row0 + wm * 64 + mt * 16 + half * 8;
            const float bn  = bias[n];
            if (MODE == 1) {
                if (col0 < D1) {                 // block-uniform scalar branch
#pragma unroll
                    for (int v = 0; v < 8; ++v) {
                        const int m = mbase + v;
                        const float val = acc[mt][nt][v] + bn +
                                          Ux[(size_t)m * D1 + n];
                        outBf[(size_t)m * NDIM + n] = f2bf(rho_f(val));
                    }
                } else {
#pragma unroll
                    for (int v = 0; v < 8; ++v) {
                        const int m = mbase + v;
                        const float val = acc[mt][nt][v] + bn;
                        outBf[(size_t)m * NDIM + n] = f2bf(rho_f(val));
                    }
                }
            } else {
#pragma unroll
                for (int v = 0; v < 8; ++v) {
                    const int m = mbase + v;
                    outF[(size_t)m * NDIM + n] = acc[mt][nt][v] + bn;
                }
            }
        }
    }
}

extern "C" void kernel_launch(void* const* d_in, const int* in_sizes, int n_in,
                              void* d_out, int out_size, void* d_ws, size_t ws_size,
                              hipStream_t stream) {
    (void)in_sizes; (void)n_in; (void)out_size; (void)ws_size;
    const float* Ux     = (const float*)d_in[0];   // [4096 x 2048]
    const float* s      = (const float*)d_in[1];   // [4096 x 4096]
    const float* Wt     = (const float*)d_in[2];   // [4096 x 4096]
    const float* b_even = (const float*)d_in[3];   // [4096]
    const float* b_odd  = (const float*)d_in[4];   // [4096]
    const float* Wm     = (const float*)d_in[5];   // [4096 x 4096]
    float* out = (float*)d_out;

    char* ws = (char*)d_ws;
    const size_t MB32 = (size_t)32 << 20;          // 4096*4096*2 bytes
    unsigned short* Wb  = (unsigned short*)(ws + 0 * MB32);
    unsigned short* Wbt = (unsigned short*)(ws + 1 * MB32);
    unsigned short* A1  = (unsigned short*)(ws + 2 * MB32);
    unsigned short* A2  = (unsigned short*)(ws + 3 * MB32);

    prep_w_kernel<<<dim3(NDIM / 32, NDIM / 32), 256, 0, stream>>>(Wt, Wm, Wb, Wbt);
    prep_a_kernel<<<16384, 256, 0, stream>>>(s, A1);

    dim3 grid(NDIM / BM, NDIM / BN);   // 32 x 16 blocks of 128x256 tiles

    // GEMM1: T = rho(s) @ W (+b_odd, +Ux, rho) -> A2 (bf16)
    //   cols < 2048 depend on k in [0,2048); cols >= 2048 on k in [0,4096)
    gemm_kernel<1><<<grid, 256, 0, stream>>>(A1, Wbt, b_odd, Ux, A2, nullptr,
                                             0, D1, 0, NDIM);

    // GEMM2: out = rho(T) @ W.T (+b_even) -> fp32
    //   cols < 2048 depend on k in [0,4096); cols >= 2048 on k in [2048,4096)
    gemm_kernel<2><<<grid, 256, 0, stream>>>(A2, Wb, b_even, nullptr, nullptr, out,
                                             0, NDIM, D1, NDIM);
}
